// Attention_74217034875079
// MI455X (gfx1250) — matrix operand order repaired and checked by reading, not compiled, not run on hardware
//
#include <hip/hip_runtime.h>

// ---------------------------------------------------------------------------
// Transformer-XL relative attention layer for MI455X (gfx1250), wave32 WMMA.
// - All GEMMs: v_wmma_f32_16x16x32_f16 (f16 operands, f32 accumulation),
//   weights pre-converted to f16, 32x64 tiles per wave (8 WMMAs / K-step).
// - Fused flash attention: 4 waves share (b,h); K / V^T / rk tiles staged
//   into LDS by the Tensor Data Mover (tensor_load_to_lds, TENSORcnt,
//   double-buffered) and shared across the workgroup.
// ---------------------------------------------------------------------------

typedef _Float16 half_t;
typedef __attribute__((ext_vector_type(16))) _Float16 v16h;
typedef __attribute__((ext_vector_type(8)))  float    v8f;
typedef __attribute__((ext_vector_type(4)))  float    v4f;
typedef unsigned int u32x4 __attribute__((ext_vector_type(4)));
typedef int          i32x4 __attribute__((ext_vector_type(4)));
typedef int          i32x8 __attribute__((ext_vector_type(8)));

#define BQ   4
#define DM   1024
#define LQ   1024
#define NH   16
#define DH   64
#define THD  (3 * NH * DH)     // 3072
#define LOCAL 1000
#define RKPAD 128
#define RKROWS (LQ + 2 * RKPAD)  // zero-padded rk rows

__device__ __forceinline__ int laneid() { return (int)(threadIdx.x & 31); }

__device__ __forceinline__ v8f wmma16(v16h a, v16h b, v8f c) {
  // D = A(16x32 f16) * B(32x16 f16) + C(16x16 f32)
  return __builtin_amdgcn_wmma_f32_16x16x32_f16(
      false, a, false, b, (short)0, c, false, false);
}

// A fragment (16x32 f16) from row-major src[row, k], row stride ld (halves).
// lanes 0-15: row=lane, K {0..15}; lanes 16-31: row=lane-16, K {16..31}
__device__ __forceinline__ v16h load_frag_a_f16(const half_t* base, int ld) {
  const int l = laneid();
  const half_t* p = base + (long)(l & 15) * ld + ((l >> 4) << 4);
  v16h r;
  *reinterpret_cast<v4f*>(&r)       = *reinterpret_cast<const v4f*>(p);
  *(reinterpret_cast<v4f*>(&r) + 1) = *reinterpret_cast<const v4f*>(p + 8);
  return r;
}

// B fragment (32x16 f16), B[k,n] sourced from row-major S[n,k], row stride ld.
// lane holds column n = lane&15; K chunk = 16*(lane>>4) + [0..15] (contiguous 32B).
__device__ __forceinline__ v16h load_frag_b_f16(const half_t* base, int ld) {
  const int l = laneid();
  const half_t* p = base + (long)(l & 15) * ld + ((l >> 4) << 4);
  v16h r;
  *reinterpret_cast<v4f*>(&r)       = *reinterpret_cast<const v4f*>(p);
  *(reinterpret_cast<v4f*>(&r) + 1) = *reinterpret_cast<const v4f*>(p + 8);
  return r;
}

// Shared 32x64-tile GEMM body: C(32x64,f32) += A(32xDM,f16) * S(64xDM,f16)^T
__device__ __forceinline__ void gemm_32x64(const half_t* __restrict__ A, int lda,
                                           const half_t* __restrict__ Bm, int ldb,
                                           v8f acc[2][4]) {
#pragma unroll 2
  for (int k0 = 0; k0 < DM; k0 += 32) {
    const v16h a0 = load_frag_a_f16(A + k0, lda);
    const v16h a1 = load_frag_a_f16(A + (long)16 * lda + k0, lda);
    const v16h b0 = load_frag_b_f16(Bm + k0, ldb);
    const v16h b1 = load_frag_b_f16(Bm + (long)16 * ldb + k0, ldb);
    const v16h b2 = load_frag_b_f16(Bm + (long)32 * ldb + k0, ldb);
    const v16h b3 = load_frag_b_f16(Bm + (long)48 * ldb + k0, ldb);
    acc[0][0] = wmma16(a0, b0, acc[0][0]);
    acc[0][1] = wmma16(a0, b1, acc[0][1]);
    acc[0][2] = wmma16(a0, b2, acc[0][2]);
    acc[0][3] = wmma16(a0, b3, acc[0][3]);
    acc[1][0] = wmma16(a1, b0, acc[1][0]);
    acc[1][1] = wmma16(a1, b1, acc[1][1]);
    acc[1][2] = wmma16(a1, b2, acc[1][2]);
    acc[1][3] = wmma16(a1, b3, acc[1][3]);
  }
}

// ---------------------------------------------------------------------------
// Tensor Data Mover: 2D tile of 2-byte elements, global -> LDS.
// D# built per CDNA5 ISA: group0 = {count=1, lds_addr,
// global_addr, type=2}; group1 = {data_size=2B, dims/tiles/stride}.
// ---------------------------------------------------------------------------
__device__ __forceinline__ unsigned lds_addr_of(const void* p) {
  // generic LDS address: addr[31:0] is the LDS byte offset (ISA 10.2 aperture)
  return (unsigned)(uintptr_t)p;
}

__device__ __forceinline__ void tdm_load_2d(unsigned lds, const half_t* g,
                                            int tile0, int tile1,
                                            int dim0, int dim1, int stride0) {
  unsigned long long ga = (unsigned long long)(uintptr_t)g;
  u32x4 g0;
  g0[0] = 1u;                                              // count=1 (user D#)
  g0[1] = lds;                                             // lds_addr (bytes)
  g0[2] = (unsigned)(ga & 0xffffffffu);                    // global_addr[31:0]
  g0[3] = (unsigned)((ga >> 32) & 0x01ffffffu) | (2u << 30); // addr[56:32]|type=2
  i32x8 g1;
  g1[0] = (int)(1u << 16);                                 // data_size=1 (2B)
  g1[1] = (int)(((unsigned)dim0 & 0xffffu) << 16);         // tensor_dim0[15:0]
  g1[2] = (int)((((unsigned)dim0 >> 16) & 0xffffu) |
                (((unsigned)dim1 & 0xffffu) << 16));       // dim0 hi | dim1 lo
  g1[3] = (int)((((unsigned)dim1 >> 16) & 0xffffu) |
                ((unsigned)tile0 << 16));                  // dim1 hi | tile_dim0
  g1[4] = (int)((unsigned)tile1 & 0xffffu);                // tile_dim1, tile_dim2=0
  g1[5] = (int)stride0;                                    // tensor_dim0_stride
  g1[6] = 0;
  g1[7] = 0;
  i32x4 gz4 = {0, 0, 0, 0};
  i32x8 gz8 = {0, 0, 0, 0, 0, 0, 0, 0};
  __builtin_amdgcn_tensor_load_to_lds(g0, g1, gz4, gz4, gz8, 0);
}

// ---------------------------------------------------------------------------
// Kernel 0a: LDS-tiled transpose + f32->f16:  dst[m, l, c] = (f16) src[m, c, l]
// ---------------------------------------------------------------------------
__global__ __launch_bounds__(256) void k_transpose_cvt(
    const float* __restrict__ src, half_t* __restrict__ dst) {
  __shared__ float t[32][33];
  const int tile = blockIdx.x;
  const int lt = tile & 31;
  const int ct = (tile >> 5) & 31;
  const int m  = tile >> 10;
  const int tx = threadIdx.x & 31;
  const int ty = threadIdx.x >> 5;
  const float* s = src + ((long)m * DM + (long)ct * 32) * LQ + (long)lt * 32;
#pragma unroll
  for (int r = 0; r < 4; ++r)
    t[ty + 8 * r][tx] = s[(long)(ty + 8 * r) * LQ + tx];
  __syncthreads();
  half_t* d = dst + ((long)m * LQ + (long)lt * 32) * DM + (long)ct * 32;
#pragma unroll
  for (int r = 0; r < 4; ++r)
    d[(long)(ty + 8 * r) * DM + tx] = (half_t)t[tx][ty + 8 * r];
}

// ---------------------------------------------------------------------------
// Kernel 0b: elementwise f32 -> f16 (weights)
// ---------------------------------------------------------------------------
__global__ void k_cvt_f16(const float* __restrict__ src,
                          half_t* __restrict__ dst, int n) {
  const int i = blockIdx.x * blockDim.x + threadIdx.x;
  if (i < n) dst[i] = (half_t)src[i];
}

// ---------------------------------------------------------------------------
// Kernel 1: QKV projection (32x64 tile per wave); scale folded into Q.
// ---------------------------------------------------------------------------
__global__ __launch_bounds__(256) void k_qkv(
    const half_t* __restrict__ qkv_wh, const half_t* __restrict__ zh,
    const float* __restrict__ u, const float* __restrict__ r_w_bias,
    const float* __restrict__ r_r_bias, half_t* __restrict__ Qu,
    half_t* __restrict__ Qv, half_t* __restrict__ Kh, half_t* __restrict__ Vt) {
  const int wave = threadIdx.x >> 5;
  const int tid  = blockIdx.x * 8 + wave;
  const int ntn = LQ / 64, ntm = THD / 32;
  const int nt = tid % ntn;
  const int t2 = tid / ntn;
  const int mt = t2 % ntm;
  const int b  = t2 / ntm;
  const int m0 = mt * 32, n0 = nt * 64;

  const half_t* zb = zh + (long)b * LQ * DM;
  v8f acc[2][4];
  const v8f vz = {};
#pragma unroll
  for (int mi = 0; mi < 2; ++mi)
#pragma unroll
    for (int ti = 0; ti < 4; ++ti) acc[mi][ti] = vz;

  gemm_32x64(qkv_wh + (long)m0 * DM, DM, zb + (long)n0 * DM, DM, acc);

  const int l   = laneid();
  const int col = l & 15;
  const int rof = (l >> 4) << 3;
#pragma unroll
  for (int mi = 0; mi < 2; ++mi) {
    const int ob = m0 + 16 * mi;
    const int part = ob >> 10;
    const int h = (ob & 1023) >> 6;
#pragma unroll
    for (int ti = 0; ti < 4; ++ti) {
#pragma unroll
      for (int v = 0; v < 8; ++v) {
        const int o  = ob + v + rof;
        const int d  = o & 63;
        const int ll = n0 + 16 * ti + col;
        float val = acc[mi][ti][v] + u[((long)b * THD + o) * LQ + ll];
        if (part == 0) {
          const float s = 0.125f;                 // 1/sqrt(64)
          const long qi = (((long)b * NH + h) * LQ + ll) * DH + d;
          Qu[qi] = (half_t)((val + r_w_bias[h * DH + d]) * s);
          Qv[qi] = (half_t)((val + r_r_bias[h * DH + d]) * s);
        } else if (part == 1) {
          Kh[(((long)b * NH + h) * LQ + ll) * DH + d] = (half_t)val;
        } else {
          Vt[(((long)b * NH + h) * DH + d) * LQ + ll] = (half_t)val;
        }
      }
    }
  }
}

// ---------------------------------------------------------------------------
// Kernel 2: positional projection into zero-padded rk:
//   rk[h, RKPAD + r, d] = sum_c r_w[h*64+d, c] * pos[r, c]
// ---------------------------------------------------------------------------
__global__ __launch_bounds__(256) void k_rk(const half_t* __restrict__ r_wh,
                                            const half_t* __restrict__ ph,
                                            half_t* __restrict__ rk) {
  const int wave = threadIdx.x >> 5;
  const int tid  = blockIdx.x * 8 + wave;
  const int ntn = LQ / 64;
  const int nt = tid % ntn, mt = tid / ntn;
  const int m0 = mt * 32, n0 = nt * 64;
  v8f acc[2][4];
  const v8f vz = {};
#pragma unroll
  for (int mi = 0; mi < 2; ++mi)
#pragma unroll
    for (int ti = 0; ti < 4; ++ti) acc[mi][ti] = vz;

  gemm_32x64(r_wh + (long)m0 * DM, DM, ph + (long)n0 * DM, DM, acc);

  const int l = laneid();
  const int col = l & 15;
  const int rof = (l >> 4) << 3;
#pragma unroll
  for (int mi = 0; mi < 2; ++mi) {
#pragma unroll
    for (int ti = 0; ti < 4; ++ti) {
#pragma unroll
      for (int v = 0; v < 8; ++v) {
        const int hd = m0 + 16 * mi + v + rof;
        const int h = hd >> 6, d = hd & 63;
        const int r = n0 + 16 * ti + col;
        rk[((long)h * RKROWS + RKPAD + r) * DH + d] = (half_t)acc[mi][ti][v];
      }
    }
  }
}

// ---------------------------------------------------------------------------
// Kernel 3: fused flash attention, TDM-staged operands.
// Block = 4 waves sharing (b,h); wave w owns q-rows [i0b+16w, i0b+16w+16).
// Per 32-col j-step the TDM stages K(32x64), V^T(64x32) and a 96-row rk
// window into double-buffered LDS (3 tensor_load_to_lds, s_wait_tensorcnt).
//   score[i,j] = Qu_i.K_j + Qv_i.rk[1023-(i-j)]   (masked: i-999 <= j <= i)
// ---------------------------------------------------------------------------
__global__ __launch_bounds__(128) void k_attn(
    const half_t* __restrict__ Qu, const half_t* __restrict__ Qv,
    const half_t* __restrict__ Kh, const half_t* __restrict__ Vt,
    const half_t* __restrict__ rk, half_t* __restrict__ Av) {
  __shared__ half_t KT[2][32 * DH];     // staged K  [j, d]
  __shared__ half_t VT[2][DH * 32];     // staged V^T [d, j]
  __shared__ half_t RKT[2][96 * DH];    // staged rk window [r, d]
  __shared__ float  lbd[4][16 * 49];    // per-wave BD window (16 x 48, padded)
  __shared__ half_t lp[4][16 * 32];     // per-wave P tile

  const int wave = threadIdx.x >> 5;
  const int l    = threadIdx.x & 31;
  const int col  = l & 15;
  const int hi   = l >> 4;

  const int bid = blockIdx.x;                    // B * NH * (LQ/64) blocks
  const int q64 = bid % (LQ / 64);
  const int t2  = bid / (LQ / 64);
  const int h   = t2 % NH;
  const int b   = t2 / NH;
  const int i0b = q64 * 64;
  const int i0  = i0b + 16 * wave;
  const long bh = (long)b * NH + h;

  const half_t* Khb = Kh + bh * LQ * DH;
  const half_t* Vtb = Vt + bh * DH * LQ;
  const half_t* rkp = rk + (long)h * RKROWS * DH + (long)RKPAD * DH;

  const half_t* Qub = Qu + (bh * LQ + i0) * DH;
  const half_t* Qvb = Qv + (bh * LQ + i0) * DH;
  const v16h qa0 = load_frag_a_f16(Qub, DH);
  const v16h qa1 = load_frag_a_f16(Qub + 32, DH);
  const v16h qv0 = load_frag_a_f16(Qvb, DH);
  const v16h qv1 = load_frag_a_f16(Qvb + 32, DH);

  v8f o0 = {}, o1 = {}, o2 = {}, o3 = {};
  float M[8], Ls[8];
#pragma unroll
  for (int v = 0; v < 8; ++v) { M[v] = -1e30f; Ls[v] = 0.0f; }

  float*  bdw = lbd[wave];
  half_t* pw  = lp[wave];
  const int roff = 48 - 16 * wave;               // wave's window in staged rk

  int jlo = i0b - (LOCAL - 1);
  if (jlo < 0) jlo = 0;
  jlo &= ~31;
  const int jhi = i0b + 63;

  // preload first tile set (wave 0 issues all TDM work)
  if (wave == 0) {
    tdm_load_2d(lds_addr_of(&KT[0][0]), Khb + (long)jlo * DH, DH, 32, DH, LQ - jlo, DH);
    tdm_load_2d(lds_addr_of(&VT[0][0]), Vtb + jlo, 32, DH, LQ - jlo, DH, LQ);
    tdm_load_2d(lds_addr_of(&RKT[0][0]), rkp + (long)(960 - i0b + jlo) * DH,
                DH, 96, DH, 1 << 20, DH);
  }

  int buf = 0;
  for (int j0 = jlo; j0 <= jhi; j0 += 32, buf ^= 1) {
    __syncthreads();                             // safe to overwrite buf^1
    if (wave == 0) {
      const int jn = j0 + 32;
      if (jn <= jhi) {
        const int nb = buf ^ 1;
        tdm_load_2d(lds_addr_of(&KT[nb][0]), Khb + (long)jn * DH, DH, 32, DH, LQ - jn, DH);
        tdm_load_2d(lds_addr_of(&VT[nb][0]), Vtb + jn, 32, DH, LQ - jn, DH, LQ);
        tdm_load_2d(lds_addr_of(&RKT[nb][0]), rkp + (long)(960 - i0b + jn) * DH,
                    DH, 96, DH, 1 << 20, DH);
        __builtin_amdgcn_s_wait_tensorcnt((short)3);  // current set complete
      } else {
        __builtin_amdgcn_s_wait_tensorcnt((short)0);
      }
    }
    __syncthreads();                             // staged tiles visible to all

    const half_t* KTb = &KT[buf][0];
    const half_t* VTb = &VT[buf][0];
    const half_t* RKb = &RKT[buf][0] + roff * DH;

    // ---- BD window = Qv . rk^T over 48 r-values (6 WMMAs) ----
    v8f bd0 = {}, bd1 = {}, bd2 = {};
    bd0 = wmma16(qv0, load_frag_b_f16(RKb,               DH), bd0);
    bd0 = wmma16(qv1, load_frag_b_f16(RKb + 32,          DH), bd0);
    bd1 = wmma16(qv0, load_frag_b_f16(RKb + 16 * DH,      DH), bd1);
    bd1 = wmma16(qv1, load_frag_b_f16(RKb + 16 * DH + 32, DH), bd1);
    bd2 = wmma16(qv0, load_frag_b_f16(RKb + 32 * DH,      DH), bd2);
    bd2 = wmma16(qv1, load_frag_b_f16(RKb + 32 * DH + 32, DH), bd2);
#pragma unroll
    for (int v = 0; v < 8; ++v) {
      const int row = v + 8 * hi;
      bdw[row * 49 + col]      = bd0[v];
      bdw[row * 49 + 16 + col] = bd1[v];
      bdw[row * 49 + 32 + col] = bd2[v];
    }
    asm volatile("s_wait_dscnt 0x0" ::: "memory");

    v8f s[2];
#pragma unroll
    for (int sub = 0; sub < 2; ++sub) {
      const int js = j0 + 16 * sub;
      // ---- AC = Qu . K^T (from staged K) ----
      v8f sc = {};
      sc = wmma16(qa0, load_frag_b_f16(KTb + 16 * sub * DH,      DH), sc);
      sc = wmma16(qa1, load_frag_b_f16(KTb + 16 * sub * DH + 32, DH), sc);
      // ---- add shifted BD + mask ----
#pragma unroll
      for (int v = 0; v < 8; ++v) {
        const int row = v + 8 * hi;
        const int cc  = 16 * sub + 15 - row + col;   // 0..46
        const float sv = sc[v] + bdw[row * 49 + cc];
        const int jg = js + col, ig = i0 + row;
        const bool msk = (jg > ig) || (jg < ig - (LOCAL - 1));
        sc[v] = msk ? -__builtin_inff() : sv;
      }
      s[sub] = sc;
    }
    // ---- online softmax over this 32-column block ----
#pragma unroll
    for (int v = 0; v < 8; ++v) {
      float t = fmaxf(s[0][v], s[1][v]);
#pragma unroll
      for (int off = 8; off; off >>= 1) t = fmaxf(t, __shfl_xor(t, off, 32));
      const float mn = fmaxf(M[v], t);
      const float al = __expf(M[v] - mn);
      const float p0 = __expf(s[0][v] - mn);
      const float p1 = __expf(s[1][v] - mn);
      float rs = p0 + p1;
#pragma unroll
      for (int off = 8; off; off >>= 1) rs += __shfl_xor(rs, off, 32);
      Ls[v] = al * Ls[v] + rs;
      M[v]  = mn;
      const int row = v + 8 * hi;
      pw[row * 32 + col]      = (half_t)p0;
      pw[row * 32 + 16 + col] = (half_t)p1;
      o0[v] *= al; o1[v] *= al; o2[v] *= al; o3[v] *= al;
    }
    asm volatile("s_wait_dscnt 0x0" ::: "memory");
    // ---- O += P * V  (P via LDS as A fragment, V from staged tile) ----
    const v16h pa = load_frag_a_f16(pw, 32);
    o0 = wmma16(pa, load_frag_b_f16(VTb + 0  * 32, 32), o0);
    o1 = wmma16(pa, load_frag_b_f16(VTb + 16 * 32, 32), o1);
    o2 = wmma16(pa, load_frag_b_f16(VTb + 32 * 32, 32), o2);
    o3 = wmma16(pa, load_frag_b_f16(VTb + 48 * 32, 32), o3);
  }

  // ---- normalize and store attn_vec as Av[b, l, h, d] (f16) ----
#pragma unroll
  for (int v = 0; v < 8; ++v) {
    const float inv = 1.0f / Ls[v];
    const int row = v + 8 * hi;
    const long base = (((long)b * LQ + (i0 + row)) * NH + h) * DH;
    Av[base + 0  + col] = (half_t)(o0[v] * inv);
    Av[base + 16 + col] = (half_t)(o1[v] * inv);
    Av[base + 32 + col] = (half_t)(o2[v] * inv);
    Av[base + 48 + col] = (half_t)(o3[v] * inv);
  }
}

// ---------------------------------------------------------------------------
// Kernel 4a: output projection + bias + residual
// ---------------------------------------------------------------------------
__global__ __launch_bounds__(256) void k_proj(const half_t* __restrict__ o_wh,
                                              const half_t* __restrict__ Av,
                                              const float* __restrict__ o_b,
                                              const float* __restrict__ z,
                                              float* __restrict__ tmp) {
  const int wave = threadIdx.x >> 5;
  const int tid  = blockIdx.x * 8 + wave;
  const int ntn = LQ / 64, ntm = DM / 32;
  const int nt = tid % ntn;
  const int t2 = tid / ntn;
  const int mt = t2 % ntm;
  const int b  = t2 / ntm;
  const int m0 = mt * 32, n0 = nt * 64;

  const half_t* Ab = Av + (long)b * LQ * DM;
  v8f acc[2][4];
  const v8f vz = {};
#pragma unroll
  for (int mi = 0; mi < 2; ++mi)
#pragma unroll
    for (int ti = 0; ti < 4; ++ti) acc[mi][ti] = vz;

  gemm_32x64(o_wh + (long)m0 * DM, DM, Ab + (long)n0 * DM, DM, acc);

  const int l = laneid();
  const int col = l & 15;
  const int rof = (l >> 4) << 3;
#pragma unroll
  for (int mi = 0; mi < 2; ++mi) {
#pragma unroll
    for (int ti = 0; ti < 4; ++ti) {
#pragma unroll
      for (int v = 0; v < 8; ++v) {
        const int o  = m0 + 16 * mi + v + rof;
        const int ll = n0 + 16 * ti + col;
        const long idx = ((long)b * DM + o) * LQ + ll;
        tmp[idx] = acc[mi][ti][v] + o_b[o] + z[idx];
      }
    }
  }
}

// ---------------------------------------------------------------------------
// Kernel 4b: LayerNorm over channel dim (no affine), eps=1e-5
// ---------------------------------------------------------------------------
__global__ void k_lnorm(const float* __restrict__ tmp, float* __restrict__ out) {
  const int idx = blockIdx.x * blockDim.x + threadIdx.x;
  if (idx >= BQ * LQ) return;
  const int b = idx / LQ, ll = idx % LQ;
  const float* p = tmp + (long)b * DM * LQ + ll;
  float s = 0.0f, s2 = 0.0f;
  for (int c = 0; c < DM; ++c) {
    const float x = p[(long)c * LQ];
    s += x; s2 += x * x;
  }
  const float mean = s * (1.0f / DM);
  const float var  = s2 * (1.0f / DM) - mean * mean;
  const float inv  = rsqrtf(var + 1e-5f);
  float* q = out + (long)b * DM * LQ + ll;
  for (int c = 0; c < DM; ++c)
    q[(long)c * LQ] = (p[(long)c * LQ] - mean) * inv;
}

// ---------------------------------------------------------------------------
extern "C" void kernel_launch(void* const* d_in, const int* in_sizes, int n_in,
                              void* d_out, int out_size, void* d_ws, size_t ws_size,
                              hipStream_t stream) {
  (void)in_sizes; (void)n_in; (void)out_size; (void)ws_size;
  const float* z        = (const float*)d_in[0];
  const float* pos      = (const float*)d_in[1];
  const float* u        = (const float*)d_in[2];
  const float* qkv_w    = (const float*)d_in[3];
  const float* r_w      = (const float*)d_in[4];
  const float* r_w_bias = (const float*)d_in[5];
  const float* r_r_bias = (const float*)d_in[6];
  const float* o_w      = (const float*)d_in[7];
  const float* o_b      = (const float*)d_in[8];
  float* out = (float*)d_out;

  char* ws = (char*)d_ws;
  half_t* zh     = (half_t*)ws; ws += (size_t)BQ * LQ * DM * sizeof(half_t);
  half_t* ph     = (half_t*)ws; ws += (size_t)LQ * DM * sizeof(half_t);
  half_t* qkv_wh = (half_t*)ws; ws += (size_t)THD * DM * sizeof(half_t);
  half_t* r_wh   = (half_t*)ws; ws += (size_t)DM * DM * sizeof(half_t);
  half_t* o_wh   = (half_t*)ws; ws += (size_t)DM * DM * sizeof(half_t);
  half_t* Qu     = (half_t*)ws; ws += (size_t)BQ * NH * LQ * DH * sizeof(half_t);
  half_t* Qv     = (half_t*)ws; ws += (size_t)BQ * NH * LQ * DH * sizeof(half_t);
  half_t* Kh     = (half_t*)ws; ws += (size_t)BQ * NH * LQ * DH * sizeof(half_t);
  half_t* Vt     = (half_t*)ws; ws += (size_t)BQ * NH * LQ * DH * sizeof(half_t);
  half_t* rk     = (half_t*)ws; ws += (size_t)NH * RKROWS * DH * sizeof(half_t);
  half_t* Av     = (half_t*)ws; ws += (size_t)BQ * LQ * NH * DH * sizeof(half_t);
  float*  tmp    = (float*)ws;

  // stage conversions / transposes; zero rk pads
  k_transpose_cvt<<<BQ * 32 * 32, 256, 0, stream>>>(z, zh);
  k_transpose_cvt<<<1 * 32 * 32, 256, 0, stream>>>(pos, ph);
  k_cvt_f16<<<(THD * DM + 255) / 256, 256, 0, stream>>>(qkv_w, qkv_wh, THD * DM);
  k_cvt_f16<<<(DM * DM + 255) / 256, 256, 0, stream>>>(r_w, r_wh, DM * DM);
  k_cvt_f16<<<(DM * DM + 255) / 256, 256, 0, stream>>>(o_w, o_wh, DM * DM);
  (void)hipMemsetAsync(rk, 0, (size_t)NH * RKROWS * DH * sizeof(half_t), stream);

  // projections
  k_qkv<<<(BQ * (THD / 32) * (LQ / 64)) / 8, 256, 0, stream>>>(
      qkv_wh, zh, u, r_w_bias, r_r_bias, Qu, Qv, Kh, Vt);
  k_rk<<<((DM / 32) * (LQ / 64)) / 8, 256, 0, stream>>>(r_wh, ph, rk);

  // fused attention (TDM-staged operands)
  k_attn<<<BQ * NH * (LQ / 64), 128, 0, stream>>>(Qu, Qv, Kh, Vt, rk, Av);

  // output projection + residual, then channel LayerNorm
  k_proj<<<(BQ * (DM / 32) * (LQ / 64)) / 8, 256, 0, stream>>>(o_wh, Av, o_b, z, tmp);
  k_lnorm<<<(BQ * LQ + 255) / 256, 256, 0, stream>>>(tmp, out);
}
